// BERT_43233140802164
// MI455X (gfx1250) — compile-verified, matrix operand order checked by reading, not run on hardware
//
#include <hip/hip_runtime.h>
#include <math.h>

typedef __attribute__((ext_vector_type(16))) __bf16 v16bf;
typedef __attribute__((ext_vector_type(8)))  __bf16 v8bf;
typedef __attribute__((ext_vector_type(8)))  float  v8f;

union FragBF { v16bf v; v8bf h[2]; };

static __device__ __forceinline__ float fast_gelu(float v) {
    const float c = 0.7978845608028654f; // sqrt(2/pi)
    float u = c * (v + 0.044715f * v * v * v);
    return 0.5f * v * (1.0f + tanhf(u));
}

// ---------------------------------------------------------------------------
// Embedding: x[b,t,:] = wte[idx[b,t],:] + wpe[t,:]   (f32 residual master)
// ---------------------------------------------------------------------------
__global__ void embed_kernel(const int* __restrict__ idx,
                             const float* __restrict__ wte,
                             const float* __restrict__ wpe,
                             float* __restrict__ x, int T, int C) {
    int row = blockIdx.x;            // b*T + t
    int t   = row % T;
    int tok = idx[row];
    const float* we = wte + (size_t)tok * C;
    const float* wp = wpe + (size_t)t * C;
    float* o = x + (size_t)row * C;
    for (int c = threadIdx.x; c < C; c += blockDim.x)
        o[c] = we[c] + wp[c];
}

// ---------------------------------------------------------------------------
// LayerNorm (C=768), f32 in -> bf16 out. One block (8 waves) per row.
// ---------------------------------------------------------------------------
__global__ __launch_bounds__(256)
void ln_kernel(const float* __restrict__ x, const float* __restrict__ w,
               const float* __restrict__ b, __bf16* __restrict__ out, int C) {
    int row = blockIdx.x;
    const float* xr = x + (size_t)row * C;
    float s = 0.f, sq = 0.f;
    for (int i = threadIdx.x; i < C; i += blockDim.x) {
        float v = xr[i];
        s += v; sq += v * v;
    }
#pragma unroll
    for (int off = 16; off; off >>= 1) {
        s  += __shfl_xor(s,  off, 32);
        sq += __shfl_xor(sq, off, 32);
    }
    __shared__ float rs[8], rq[8];
    int wid = threadIdx.x >> 5;
    if ((threadIdx.x & 31) == 0) { rs[wid] = s; rq[wid] = sq; }
    __syncthreads();
    s = 0.f; sq = 0.f;
#pragma unroll
    for (int i = 0; i < 8; ++i) { s += rs[i]; sq += rq[i]; }
    float mean = s / (float)C;
    float var  = sq / (float)C - mean * mean;
    float rstd = rsqrtf(var + 1e-5f);
    __bf16* orow = out + (size_t)row * C;
    for (int i = threadIdx.x; i < C; i += blockDim.x)
        orow[i] = (__bf16)((xr[i] - mean) * rstd * w[i] + b[i]);
}

// ---------------------------------------------------------------------------
// GEMM: out[M,N] = A_bf16[M,K] * W_f32[N,K]^T (+bias)(+GELU)(+res_f32)
// Block tile 128x128x32, 8 waves x (64x32). Double-buffered LDS; A-tiles are
// moved with GLOBAL_LOAD_ASYNC_TO_LDS_B128 (ASYNCcnt) overlapped with WMMA.
// ---------------------------------------------------------------------------
#define BM 128
#define BN 128
#define BK 32
#define BKP 40   // pad: row stride 80B keeps 16B-aligned fragment loads

__global__ __launch_bounds__(256)
void gemm_bf16_kernel(const __bf16* __restrict__ A,   // [M,K] bf16
                      const float* __restrict__ W,    // [N,K] f32
                      const float* __restrict__ bias, // [N] or null
                      const float* __restrict__ res,  // [M,N] f32 or null
                      void* __restrict__ out,         // [M,N]
                      int M, int N, int K, int gelu, int out_bf16) {
    __shared__ __bf16 As[2][BM][BKP];
    __shared__ __bf16 Bs[2][BN][BKP];

    const int tid  = threadIdx.x;
    const int lane = tid & 31;
    const int wid  = tid >> 5;
    const int half = lane >> 4;
    const int r    = lane & 15;
    const int wm   = wid >> 2;        // 0..1
    const int wn   = wid & 3;         // 0..3
    const int bm   = blockIdx.y * BM;
    const int bn   = blockIdx.x * BN;

    const int srow = tid >> 1;        // 0..127
    const int scol = (tid & 1) * 16;  // element 0 or 16

    const uint32_t ldsA0 = (uint32_t)(uintptr_t)&As[0][0][0];

    const v8f vzero = {0.f,0.f,0.f,0.f,0.f,0.f,0.f,0.f};
    v8f acc[4][2];
#pragma unroll
    for (int i = 0; i < 4; ++i)
#pragma unroll
        for (int j = 0; j < 2; ++j) acc[i][j] = vzero;

    const int nk = K / BK;

    // --- tile staging helpers -------------------------------------------
    auto issueA = [&](int buf, int kk) {
        // async copy 16 bf16 (two B128 ops) of A into LDS, no VGPR roundtrip
        uint64_t g = (uint64_t)(uintptr_t)(A + (size_t)(bm + srow) * K + kk + scol);
        uint32_t l = ldsA0 + (uint32_t)(((buf * BM + srow) * BKP + scol) * 2);
        asm volatile("global_load_async_to_lds_b128 %0, %1, off"
                     :: "v"(l), "v"(g) : "memory");
        asm volatile("global_load_async_to_lds_b128 %0, %1, off"
                     :: "v"(l + 16u), "v"(g + 16ull) : "memory");
    };
    auto stageB = [&](int buf, int kk) {
        int nrow = bn + srow;
        if (nrow < N) {
            const float* src = W + (size_t)nrow * K + kk + scol;
#pragma unroll
            for (int u = 0; u < 4; ++u) {
                float4 f = *(const float4*)(src + u * 4);
                Bs[buf][srow][scol + u*4 + 0] = (__bf16)f.x;
                Bs[buf][srow][scol + u*4 + 1] = (__bf16)f.y;
                Bs[buf][srow][scol + u*4 + 2] = (__bf16)f.z;
                Bs[buf][srow][scol + u*4 + 3] = (__bf16)f.w;
            }
        } else {
#pragma unroll
            for (int e = 0; e < 16; ++e) Bs[buf][srow][scol + e] = (__bf16)0.f;
        }
    };

    // --- software pipeline ----------------------------------------------
    issueA(0, 0);
    stageB(0, 0);
    for (int k = 0; k < nk; ++k) {
        int cur = k & 1, nxt = cur ^ 1;
        if (k + 1 < nk) {
            issueA(nxt, (k + 1) * BK);
            stageB(nxt, (k + 1) * BK);
            // tile k's 2 copies done; tile k+1's 2 remain in flight
            asm volatile("s_wait_asynccnt 0x2" ::: "memory");
        } else {
            asm volatile("s_wait_asynccnt 0x0" ::: "memory");
        }
        __syncthreads();

        FragBF a[4], bf[2];
#pragma unroll
        for (int i = 0; i < 4; ++i) {
            int rr = wm * 64 + i * 16 + r;
            a[i].h[0] = *(const v8bf*)&As[cur][rr][half * 8];
            a[i].h[1] = *(const v8bf*)&As[cur][rr][16 + half * 8];
        }
#pragma unroll
        for (int j = 0; j < 2; ++j) {
            int rr = wn * 32 + j * 16 + r;
            bf[j].h[0] = *(const v8bf*)&Bs[cur][rr][half * 8];
            bf[j].h[1] = *(const v8bf*)&Bs[cur][rr][16 + half * 8];
        }
#pragma unroll
        for (int i = 0; i < 4; ++i)
#pragma unroll
            for (int j = 0; j < 2; ++j)
                acc[i][j] = __builtin_amdgcn_wmma_f32_16x16x32_bf16(
                    false, a[i].v, false, bf[j].v, (short)0, acc[i][j],
                    false, false);
        __syncthreads();
    }

    // --- epilogue: lane (half,r), vgpr e -> row e+8*half, col r ----------
    float*  outf = (float*)out;
    __bf16* outb = (__bf16*)out;
#pragma unroll
    for (int i = 0; i < 4; ++i) {
#pragma unroll
        for (int j = 0; j < 2; ++j) {
#pragma unroll
            for (int e = 0; e < 8; ++e) {
                int row = bm + wm * 64 + i * 16 + e + 8 * half;
                int col = bn + wn * 32 + j * 16 + r;
                if (col < N) {
                    float v = acc[i][j][e];
                    if (bias) v += bias[col];
                    if (gelu) v = fast_gelu(v);
                    size_t o = (size_t)row * N + col;
                    if (res) v += res[o];
                    if (out_bf16) outb[o] = (__bf16)v;
                    else          outf[o] = v;
                }
            }
        }
    }
    (void)M;
}

// ---------------------------------------------------------------------------
// Flash attention, D=64, non-causal, bf16 in/out, f32 online softmax.
// Block = 128 thr (4 waves) handles 64 query rows of one (b,h).
// ---------------------------------------------------------------------------
#define FQ 64
#define FKV 64
#define FP 72   // padded row: 144B stride, 16B aligned fragments

__global__ __launch_bounds__(128)
void flash_attn_kernel(const __bf16* __restrict__ qkv,  // [B*T, 3C]
                       __bf16* __restrict__ y,          // [B*T, C]
                       int T, int C3, int C) {
    const int bq = blockIdx.x;
    const int h  = blockIdx.y;
    const int b  = blockIdx.z;

    __shared__ __bf16 Qs[FQ][FP];
    __shared__ __bf16 Ks[FKV][FP];
    __shared__ __bf16 Vt[64][FP];        // V transposed: [d][kv]
    __shared__ __bf16 Ps[4][16][FP];     // per-wave P tile

    const int tid  = threadIdx.x;
    const int lane = tid & 31;
    const int wid  = tid >> 5;
    const int half = lane >> 4;
    const int r    = lane & 15;

    const float scale = 0.125f;          // 1/sqrt(64)
    const __bf16* qbase = qkv + (size_t)b * T * C3 + h * 64;
    const __bf16* kbase = qbase + C;
    const __bf16* vbase = qbase + 2 * C;

    // stage Q block (raw bf16 copy; scale applied post-WMMA)
    {
        int qr = tid >> 1;
        int cb = (tid & 1) * 32;
        const __bf16* src = qbase + (size_t)(bq * FQ + qr) * C3 + cb;
#pragma unroll
        for (int u = 0; u < 4; ++u)
            *(v8bf*)&Qs[qr][cb + u*8] = *(const v8bf*)(src + u*8);
    }
    __syncthreads();

    const int qrb = wid * 16;
    FragBF qf[2];
#pragma unroll
    for (int kc = 0; kc < 2; ++kc) {
        qf[kc].h[0] = *(const v8bf*)&Qs[qrb + r][kc*32 + half*8];
        qf[kc].h[1] = *(const v8bf*)&Qs[qrb + r][kc*32 + 16 + half*8];
    }

    const v8f vzero = {0.f,0.f,0.f,0.f,0.f,0.f,0.f,0.f};
    v8f acc[4];
#pragma unroll
    for (int nt = 0; nt < 4; ++nt) acc[nt] = vzero;
    float m_run[8], l_run[8];
#pragma unroll
    for (int j = 0; j < 8; ++j) { m_run[j] = -1e30f; l_run[j] = 0.f; }

    const int nkv = T / FKV;
    for (int it = 0; it < nkv; ++it) {
        __syncthreads();
        // stage K (copy) and V^T (scatter transpose)
        {
            int kr = tid >> 1;
            int cb = (tid & 1) * 32;
            const __bf16* ks = kbase + (size_t)(it * FKV + kr) * C3 + cb;
#pragma unroll
            for (int u = 0; u < 4; ++u)
                *(v8bf*)&Ks[kr][cb + u*8] = *(const v8bf*)(ks + u*8);
            const __bf16* vs = vbase + (size_t)(it * FKV + kr) * C3 + cb;
#pragma unroll
            for (int u = 0; u < 4; ++u) {
                v8bf f = *(const v8bf*)(vs + u*8);
#pragma unroll
                for (int e = 0; e < 8; ++e)
                    Vt[cb + u*8 + e][kr] = f[e];
            }
        }
        __syncthreads();

        // S = Q K^T  (16 x 64 per wave)
        v8f s[4];
#pragma unroll
        for (int nt = 0; nt < 4; ++nt) {
            v8f sa = vzero;
#pragma unroll
            for (int kc = 0; kc < 2; ++kc) {
                FragBF kf;
                kf.h[0] = *(const v8bf*)&Ks[nt*16 + r][kc*32 + half*8];
                kf.h[1] = *(const v8bf*)&Ks[nt*16 + r][kc*32 + 16 + half*8];
                sa = __builtin_amdgcn_wmma_f32_16x16x32_bf16(
                    false, qf[kc].v, false, kf.v, (short)0, sa, false, false);
            }
            s[nt] = sa;
        }
#pragma unroll
        for (int nt = 0; nt < 4; ++nt)
#pragma unroll
            for (int j = 0; j < 8; ++j) s[nt][j] *= scale;

        // online softmax: each row lives across 16 lanes of one half-wave
        float alpha[8];
#pragma unroll
        for (int j = 0; j < 8; ++j) {
            float mx = -1e30f;
#pragma unroll
            for (int nt = 0; nt < 4; ++nt) mx = fmaxf(mx, s[nt][j]);
#pragma unroll
            for (int d = 8; d; d >>= 1) mx = fmaxf(mx, __shfl_xor(mx, d, 32));
            float mnew = fmaxf(m_run[j], mx);
            float al = __expf(m_run[j] - mnew);
            float rsum = 0.f;
#pragma unroll
            for (int nt = 0; nt < 4; ++nt) {
                float p = __expf(s[nt][j] - mnew);
                s[nt][j] = p;
                rsum += p;
            }
#pragma unroll
            for (int d = 8; d; d >>= 1) rsum += __shfl_xor(rsum, d, 32);
            l_run[j] = l_run[j] * al + rsum;
            m_run[j] = mnew;
            alpha[j] = al;
        }
#pragma unroll
        for (int nt = 0; nt < 4; ++nt)
#pragma unroll
            for (int j = 0; j < 8; ++j) acc[nt][j] *= alpha[j];

        // P -> per-wave LDS tile (C-layout -> A-layout transpose via LDS)
#pragma unroll
        for (int nt = 0; nt < 4; ++nt)
#pragma unroll
            for (int j = 0; j < 8; ++j)
                Ps[wid][j + 8*half][nt*16 + r] = (__bf16)s[nt][j];
        asm volatile("s_wait_dscnt 0x0" ::: "memory");   // wave-local fence

        // O += P * V
#pragma unroll
        for (int kc = 0; kc < 2; ++kc) {
            FragBF pf;
            pf.h[0] = *(const v8bf*)&Ps[wid][r][kc*32 + half*8];
            pf.h[1] = *(const v8bf*)&Ps[wid][r][kc*32 + 16 + half*8];
#pragma unroll
            for (int nt = 0; nt < 4; ++nt) {
                FragBF vf;
                vf.h[0] = *(const v8bf*)&Vt[nt*16 + r][kc*32 + half*8];
                vf.h[1] = *(const v8bf*)&Vt[nt*16 + r][kc*32 + 16 + half*8];
                acc[nt] = __builtin_amdgcn_wmma_f32_16x16x32_bf16(
                    false, pf.v, false, vf.v, (short)0, acc[nt], false, false);
            }
        }
    }

    // write O / l  (bf16)
#pragma unroll
    for (int nt = 0; nt < 4; ++nt) {
#pragma unroll
        for (int j = 0; j < 8; ++j) {
            int q = bq * FQ + qrb + j + 8 * half;
            int d = nt * 16 + r;
            y[(size_t)(b * T + q) * C + h * 64 + d] =
                (__bf16)(acc[nt][j] / l_run[j]);
        }
    }
}

// ---------------------------------------------------------------------------
// Host-side orchestration
// ---------------------------------------------------------------------------
extern "C" void kernel_launch(void* const* d_in, const int* in_sizes, int n_in,
                              void* d_out, int out_size, void* d_ws, size_t ws_size,
                              hipStream_t stream) {
    (void)in_sizes; (void)n_in; (void)out_size; (void)ws_size;
    const int L = 12, Bb = 4, T = 1024, C = 768, H = 12, V = 30522;
    const int BT = Bb * T;

    const int*   idx    = (const int*)d_in[0];
    const float* wte    = (const float*)d_in[1];
    const float* wpe    = (const float*)d_in[2];
    const float* ln1w   = (const float*)d_in[3];
    const float* ln1b   = (const float*)d_in[4];
    const float* attnw  = (const float*)d_in[5];
    const float* attnb  = (const float*)d_in[6];
    const float* projw  = (const float*)d_in[7];
    const float* projb  = (const float*)d_in[8];
    const float* ln2w   = (const float*)d_in[9];
    const float* ln2b   = (const float*)d_in[10];
    const float* fcw    = (const float*)d_in[11];
    const float* fcbias = (const float*)d_in[12];
    const float* mprojw = (const float*)d_in[13];
    const float* mprojb = (const float*)d_in[14];
    const float* lnfw   = (const float*)d_in[15];
    const float* lnfb   = (const float*)d_in[16];
    const float* lmw    = (const float*)d_in[17];
    float* out = (float*)d_out;

    // workspace partition: f32 residual master + bf16 activations
    char* p = (char*)d_ws;
    float*  x   = (float*)p;              p += (size_t)BT * C * 4;   // [BT,C] f32
    __bf16* xn  = (__bf16*)p;             p += (size_t)BT * C * 2;   // [BT,C]
    __bf16* qkv = (__bf16*)p;             p += (size_t)BT * 3 * C * 2;
    __bf16* yb  = (__bf16*)p;             p += (size_t)BT * C * 2;
    __bf16* fcb = (__bf16*)p;             p += (size_t)BT * 4 * C * 2;

    dim3 blk256(256), blk128(128);

    embed_kernel<<<dim3(BT), blk256, 0, stream>>>(idx, wte, wpe, x, T, C);

    for (int l = 0; l < L; ++l) {
        const float* l1w = ln1w + (size_t)l * C;
        const float* l1b = ln1b + (size_t)l * C;
        const float* aw  = attnw + (size_t)l * 3 * C * C;
        const float* ab  = attnb + (size_t)l * 3 * C;
        const float* pw  = projw + (size_t)l * C * C;
        const float* pb  = projb + (size_t)l * C;
        const float* l2w = ln2w + (size_t)l * C;
        const float* l2b = ln2b + (size_t)l * C;
        const float* fw  = fcw + (size_t)l * 4 * C * C;
        const float* fb  = fcbias + (size_t)l * 4 * C;
        const float* mw  = mprojw + (size_t)l * C * 4 * C;
        const float* mb  = mprojb + (size_t)l * C;

        ln_kernel<<<dim3(BT), blk256, 0, stream>>>(x, l1w, l1b, xn, C);
        gemm_bf16_kernel<<<dim3((3*C)/BN, BT/BM), blk256, 0, stream>>>(
            xn, aw, ab, nullptr, qkv, BT, 3*C, C, 0, 1);
        flash_attn_kernel<<<dim3(T/FQ, H, Bb), blk128, 0, stream>>>(
            qkv, yb, T, 3*C, C);
        gemm_bf16_kernel<<<dim3(C/BN, BT/BM), blk256, 0, stream>>>(
            yb, pw, pb, x, x, BT, C, C, 0, 0);
        ln_kernel<<<dim3(BT), blk256, 0, stream>>>(x, l2w, l2b, xn, C);
        gemm_bf16_kernel<<<dim3((4*C)/BN, BT/BM), blk256, 0, stream>>>(
            xn, fw, fb, nullptr, fcb, BT, 4*C, C, 1, 1);
        gemm_bf16_kernel<<<dim3(C/BN, BT/BM), blk256, 0, stream>>>(
            fcb, mw, mb, x, x, BT, C, 4*C, 0, 0);
    }

    ln_kernel<<<dim3(BT), blk256, 0, stream>>>(x, lnfw, lnfb, xn, C);
    gemm_bf16_kernel<<<dim3((V + BN - 1)/BN, BT/BM), blk256, 0, stream>>>(
        xn, lmw, nullptr, nullptr, out, BT, V, C, 0, 0);
}
